// DefConv_50586124812879
// MI455X (gfx1250) — matrix operand order repaired and checked by reading, not compiled, not run on hardware
//
#include <hip/hip_runtime.h>

// ---------------------------------------------------------------------------
// Deformable conv (K=3, pad=1, stride=1) + GroupNorm(32) + ReLU for MI455X
// B=4, C_in=C_out=256, H=W=64.  Main contraction as implicit GEMM
// M=256, N=16384, K=2304 with v_wmma_f32_16x16x32_f16.  Bilinear gather is
// fused into the B-tile fill (x stays L2-resident).  BM=256 (full M) so each
// corner is gathered exactly once; BN=64 keeps 256 workgroups in flight.
// LDS tiles are double-buffered to hide gather latency under the WMMAs.
// ---------------------------------------------------------------------------

typedef __attribute__((ext_vector_type(16))) _Float16 v16h;
typedef __attribute__((ext_vector_type(8)))  float    v8f;
typedef __attribute__((ext_vector_type(4)))  float    f32x4;
typedef __attribute__((ext_vector_type(4)))  int      i32x4;

#define BATCH   4
#define CIN     256
#define COUT    256
#define HH      64
#define WW      64
#define HWSZ    4096          // 64*64
#define PP      9
#define KTOT    2304          // CIN * PP, k = p*256 + c
#define NTOT    16384         // BATCH * HWSZ

// workspace layout (bytes), all 256B aligned
#define WS_AG    0            // f16 [256][2304] reordered weights   (1,179,648 B)
#define WS_OFF   1179648      // f32 [4][18][4096] offset-conv out   (1,179,648 B)
#define WS_META  2359296      // 32B/entry * 4*9*4096                (4,718,592 B)
#define WS_Y     7077888      // f32 [256][16384] pre-GN GEMM out    (16,777,216 B)

union FragU { f32x4 f[2]; v16h h; };
union PackU { _Float16 h[8]; f32x4 f; };

// ---------------------------------------------------------------------------
// 1) Reorder conv weights w[o][c][ky][kx] (f32) -> A[o][p*256+c] (f16)
// ---------------------------------------------------------------------------
__global__ __launch_bounds__(256) void prep_w_kernel(const float* __restrict__ w,
                                                     _Float16* __restrict__ Ag) {
  int idx = blockIdx.x * 256 + threadIdx.x;      // 589824 total
  int m = idx / KTOT;
  int r = idx - m * KTOT;
  int p = r >> 8;
  int c = r & 255;
  Ag[idx] = (_Float16)w[((size_t)(m * CIN + c)) * PP + p];
}

// ---------------------------------------------------------------------------
// 2) Offset conv: off[b][oc][h][w] = sum_{c,kh,kw} x * w_off + b_off
//    grid = B*H blocks, 256 threads = 64 cols x 4 ci-quarters, 18 accs/thread
// ---------------------------------------------------------------------------
__global__ __launch_bounds__(256) void off_conv_kernel(const float* __restrict__ x,
                                                       const float* __restrict__ w_off,
                                                       const float* __restrict__ b_off,
                                                       float* __restrict__ off) {
  __shared__ float red[4 * 18 * 64];
  const int t  = threadIdx.x;
  const int b  = blockIdx.x >> 6;
  const int h  = blockIdx.x & 63;
  const int wc = t & 63;
  const int q  = t >> 6;

  float acc[18];
#pragma unroll
  for (int oc = 0; oc < 18; ++oc) acc[oc] = 0.f;

  const float* xb = x + ((size_t)b * CIN) * HWSZ;
  for (int ci = q * 64; ci < q * 64 + 64; ++ci) {
#pragma unroll
    for (int kh = 0; kh < 3; ++kh) {
      int yy  = h + kh - 1;
      bool rok = (yy >= 0) && (yy < HH);
      int yyc = yy < 0 ? 0 : (yy > 63 ? 63 : yy);
      const float* xrow = xb + ((size_t)ci << 12) + yyc * WW;
#pragma unroll
      for (int kw = 0; kw < 3; ++kw) {
        int xx = wc + kw - 1;
        float v = 0.f;
        if (rok && xx >= 0 && xx < WW) v = xrow[xx];
        const float* wp = w_off + ci * 9 + kh * 3 + kw;  // wave-uniform
#pragma unroll
        for (int oc = 0; oc < 18; ++oc) acc[oc] += v * wp[(size_t)oc * KTOT];
      }
    }
  }
#pragma unroll
  for (int oc = 0; oc < 18; ++oc) red[(q * 18 + oc) * 64 + wc] = acc[oc];
  __syncthreads();
  for (int idx = t; idx < 18 * 64; idx += 256) {
    int oc = idx >> 6, cc = idx & 63;
    float s = red[idx] + red[1152 + idx] + red[2304 + idx] + red[3456 + idx] + b_off[oc];
    off[(((size_t)b * 18 + oc) << 12) + (h << 6) + cc] = s;
  }
}

// ---------------------------------------------------------------------------
// 3) Bilinear metadata per (b,p,hw): clipped corner indices + masked weights
//    entry = 32B: int4 {i00,i01,i10,i11}; float4 {w00,w01,w10,w11}
// ---------------------------------------------------------------------------
__global__ __launch_bounds__(256) void meta_kernel(const float* __restrict__ off,
                                                   char* __restrict__ meta) {
  int idx = blockIdx.x * 256 + threadIdx.x;      // 4*9*4096 = 147456 total
  int b  = idx / (PP * HWSZ);
  int r  = idx - b * (PP * HWSZ);
  int p  = r >> 12;
  int hw = r & 4095;
  int h  = hw >> 6, wc = hw & 63;

  float dy = off[(((size_t)b * 18 + 2 * p) << 12) + hw];
  float dx = off[(((size_t)b * 18 + 2 * p + 1) << 12) + hw];
  float ys = (float)(h - 1 + p / 3) + dy;
  float xs = (float)(wc - 1 + p % 3) + dx;

  float fy0 = floorf(ys), fx0 = floorf(xs);
  int y0 = (int)fy0, x0 = (int)fx0;
  int y1 = y0 + 1,  x1 = x0 + 1;
  float wy1 = ys - fy0, wx1 = xs - fx0;
  float wy0 = 1.f - wy1, wx0 = 1.f - wx1;

  bool vy0 = (y0 >= 0) & (y0 < HH), vy1 = (y1 >= 0) & (y1 < HH);
  bool vx0 = (x0 >= 0) & (x0 < WW), vx1 = (x1 >= 0) & (x1 < WW);
  int cy0 = y0 < 0 ? 0 : (y0 > 63 ? 63 : y0);
  int cy1 = y1 < 0 ? 0 : (y1 > 63 ? 63 : y1);
  int cx0 = x0 < 0 ? 0 : (x0 > 63 ? 63 : x0);
  int cx1 = x1 < 0 ? 0 : (x1 > 63 ? 63 : x1);

  i32x4 id; id.x = cy0 * WW + cx0; id.y = cy0 * WW + cx1;
            id.z = cy1 * WW + cx0; id.w = cy1 * WW + cx1;
  f32x4 wg; wg.x = (vy0 && vx0) ? wy0 * wx0 : 0.f;
            wg.y = (vy0 && vx1) ? wy0 * wx1 : 0.f;
            wg.z = (vy1 && vx0) ? wy1 * wx0 : 0.f;
            wg.w = (vy1 && vx1) ? wy1 * wx1 : 0.f;

  char* me = meta + (size_t)idx * 32;
  *(i32x4*)me        = id;
  *(f32x4*)(me + 16) = wg;
}

// ---------------------------------------------------------------------------
// 4) Fused gather + GEMM:  y[m][n] = sum_k A[m][k] * cols[k][n] + bias[m]
//    BM=256 (full M, each corner gathered once), BN=64, KT=32 per step.
//    8 waves: 4 M-subtiles x 2 N-subtiles, wave tile 64x32 -> 8 WMMAs/step.
//    Double-buffered LDS: prefetch tile kt+1 into regs, compute kt, store.
// ---------------------------------------------------------------------------
__global__ __launch_bounds__(256) void gemm_kernel(const float* __restrict__ x,
                                                   const _Float16* __restrict__ Ag,
                                                   const char* __restrict__ meta,
                                                   const float* __restrict__ bias,
                                                   float* __restrict__ yout) {
  __shared__ _Float16 Als[2][256 * 40];   // [buf][row 0..255][k 0..31], stride 40
  __shared__ _Float16 Bls[2][64 * 40];    // [buf][pixel 0..63][k 0..31]

  const int t     = threadIdx.x;
  const int nBase = blockIdx.x * 64;

  // B-fill role: 4 channel-groups x 64 pixels
  const int g  = t >> 6, nl = t & 63;
  const int n  = nBase + nl;
  const int bb = n >> 12;
  const int hw = n & 4095;

  // compute role
  const int wv = t >> 5, lane = t & 31;
  const int wm = wv & 3, wn = wv >> 2;
  const int half = lane >> 4, l15 = lane & 15;

  const _Float16* Asrc = Ag + (size_t)t * KTOT;                       // row t
  const char* mbase = meta + ((((size_t)bb * PP) << 12) + hw) * 32;   // p=0

  v8f acc[4][2];
  const v8f vz = {0.f, 0.f, 0.f, 0.f, 0.f, 0.f, 0.f, 0.f};
#pragma unroll
  for (int mi = 0; mi < 4; ++mi)
#pragma unroll
    for (int ni = 0; ni < 2; ++ni) acc[mi][ni] = vz;

  i32x4 id = *(const i32x4*)mbase;
  f32x4 wg = *(const f32x4*)(mbase + 16);

  // ---- prologue: fill tile kt=0 into buffer 0 ----
  {
    const f32x4* s = (const f32x4*)Asrc;
    f32x4* d = (f32x4*)(Als[0] + t * 40);
    d[0] = s[0]; d[1] = s[1]; d[2] = s[2]; d[3] = s[3];

    const float* xp = x + (((size_t)bb * CIN + g * 8) << 12);
    PackU u;
#pragma unroll
    for (int i = 0; i < 8; ++i) {
      float v = wg.x * xp[id.x] + wg.y * xp[id.y] +
                wg.z * xp[id.z] + wg.w * xp[id.w];
      u.h[i] = (_Float16)v;
      xp += HWSZ;
    }
    *(f32x4*)(Bls[0] + nl * 40 + g * 8) = u.f;
  }
  __syncthreads();

  for (int kt = 0; kt < 72; ++kt) {
    const int cur = kt & 1;
    const int nx  = cur ^ 1;
    const bool hn = (kt + 1) < 72;

    // ---- issue prefetch of tile kt+1 into registers ----
    f32x4 a0, a1, a2, a3;
    float gv[8];
    if (hn) {
      const f32x4* s = (const f32x4*)(Asrc + ((kt + 1) << 5));
      a0 = s[0]; a1 = s[1]; a2 = s[2]; a3 = s[3];
      if (((kt + 1) & 7) == 0) {                 // crossing a p boundary
        const char* me = mbase + (size_t)((kt + 1) >> 3) * (HWSZ * 32);
        id = *(const i32x4*)me;
        wg = *(const f32x4*)(me + 16);
      }
      const int cb = (((kt + 1) & 7) << 5) + g * 8;
      const float* xp = x + (((size_t)bb * CIN + cb) << 12);
#pragma unroll
      for (int i = 0; i < 8; ++i) {
        gv[i] = wg.x * xp[id.x] + wg.y * xp[id.y] +
                wg.z * xp[id.z] + wg.w * xp[id.w];
        xp += HWSZ;
      }
    }

    // ---- compute tile kt from buffer cur ----
    v16h af[4], bf[2];
#pragma unroll
    for (int mi = 0; mi < 4; ++mi) {
      const f32x4* Ar = (const f32x4*)(Als[cur] + (wm * 64 + mi * 16 + l15) * 40);
      FragU u;
      u.f[0] = Ar[half];        // k = half*8    .. +7
      u.f[1] = Ar[2 + half];    // k = 16+half*8 .. +7
      af[mi] = u.h;
    }
#pragma unroll
    for (int ni = 0; ni < 2; ++ni) {
      const f32x4* Br = (const f32x4*)(Bls[cur] + (wn * 32 + ni * 16 + l15) * 40);
      FragU u;
      u.f[0] = Br[half];          // k = half*8    .. +7
      u.f[1] = Br[2 + half];      // k = 16+half*8 .. +7
      bf[ni] = u.h;
    }
#pragma unroll
    for (int mi = 0; mi < 4; ++mi)
#pragma unroll
      for (int ni = 0; ni < 2; ++ni)
        acc[mi][ni] = __builtin_amdgcn_wmma_f32_16x16x32_f16(
            false, af[mi], false, bf[ni], (short)0, acc[mi][ni], false, false);

    // ---- drain prefetch into the other buffer ----
    if (hn) {
      f32x4* d = (f32x4*)(Als[nx] + t * 40);
      d[0] = a0; d[1] = a1; d[2] = a2; d[3] = a3;
      PackU u;
#pragma unroll
      for (int i = 0; i < 8; ++i) u.h[i] = (_Float16)gv[i];
      *(f32x4*)(Bls[nx] + nl * 40 + g * 8) = u.f;
    }
    __syncthreads();
  }

  // ---- epilogue: D layout lanes0-15 M=r, lanes16-31 M=r+8; N = lane&15 ----
#pragma unroll
  for (int mi = 0; mi < 4; ++mi)
#pragma unroll
    for (int ni = 0; ni < 2; ++ni)
#pragma unroll
      for (int r = 0; r < 8; ++r) {
        int m  = wm * 64 + mi * 16 + half * 8 + r;
        int nn = nBase + wn * 32 + ni * 16 + l15;
        yout[(size_t)m * NTOT + nn] = acc[mi][ni][r] + bias[m];
      }
}

// ---------------------------------------------------------------------------
// 5) GroupNorm(32) + affine + ReLU.  One block per (b, group): 8 ch x 4096.
// ---------------------------------------------------------------------------
__global__ __launch_bounds__(256) void gn_kernel(const float* __restrict__ y,
                                                 const float* __restrict__ gamma,
                                                 const float* __restrict__ beta,
                                                 float* __restrict__ out) {
  __shared__ float s1[256];
  __shared__ float s2[256];
  const int t  = threadIdx.x;
  const int b  = blockIdx.x >> 5;
  const int gr = blockIdx.x & 31;
  const int m0 = gr * 8;

  float s = 0.f, ss = 0.f;
#pragma unroll
  for (int j = 0; j < 8; ++j) {
    const float* rowp = y + (size_t)(m0 + j) * NTOT + ((size_t)b << 12);
    for (int k = t; k < HWSZ; k += 256) {
      float v = rowp[k];
      s += v;
      ss += v * v;
    }
  }
  s1[t] = s; s2[t] = ss;
  __syncthreads();
  for (int st = 128; st > 0; st >>= 1) {
    if (t < st) { s1[t] += s1[t + st]; s2[t] += s2[t + st]; }
    __syncthreads();
  }
  const float inv = 1.f / 32768.f;
  float mu  = s1[0] * inv;
  float var = s2[0] * inv - mu * mu;
  float rsg = rsqrtf(var + 1e-5f);

#pragma unroll
  for (int j = 0; j < 8; ++j) {
    int m = m0 + j;
    float a  = gamma[m] * rsg;
    float bc = beta[m] - mu * a;
    const float* rowp = y + (size_t)m * NTOT + ((size_t)b << 12);
    float* op = out + (((size_t)b * COUT + m) << 12);
    for (int k = t; k < HWSZ; k += 256) {
      float v = rowp[k] * a + bc;
      op[k] = v > 0.f ? v : 0.f;
    }
  }
}

// ---------------------------------------------------------------------------
extern "C" void kernel_launch(void* const* d_in, const int* in_sizes, int n_in,
                              void* d_out, int out_size, void* d_ws, size_t ws_size,
                              hipStream_t stream) {
  const float* x     = (const float*)d_in[0];
  const float* w_off = (const float*)d_in[1];
  const float* b_off = (const float*)d_in[2];
  const float* w     = (const float*)d_in[3];
  const float* bias  = (const float*)d_in[4];
  const float* gamma = (const float*)d_in[5];
  const float* beta  = (const float*)d_in[6];
  float* out = (float*)d_out;

  char* ws = (char*)d_ws;
  _Float16* Ag  = (_Float16*)(ws + WS_AG);
  float*    off = (float*)(ws + WS_OFF);
  char*     meta = ws + WS_META;
  float*    y   = (float*)(ws + WS_Y);

  prep_w_kernel  <<<COUT * KTOT / 256, 256, 0, stream>>>(w, Ag);
  off_conv_kernel<<<BATCH * HH,        256, 0, stream>>>(x, w_off, b_off, off);
  meta_kernel    <<<BATCH * PP * HWSZ / 256, 256, 0, stream>>>(off, meta);
  gemm_kernel    <<<NTOT / 64,         256, 0, stream>>>(x, Ag, meta, bias, y);
  gn_kernel      <<<BATCH * 32,        256, 0, stream>>>(y, gamma, beta, out);
}
